// GraphMAE_17093969838150
// MI455X (gfx1250) — compile-verified, matrix operand order checked by reading, not run on hardware
//
#include <hip/hip_runtime.h>
#include <hip/hip_bf16.h>

typedef __attribute__((ext_vector_type(16))) _Float16 v16h;
typedef __attribute__((ext_vector_type(8)))  _Float16 v8h;
typedef __attribute__((ext_vector_type(8)))  float    v8f;
typedef __attribute__((ext_vector_type(4)))  float    v4f;

#define N_NODES  50000
#define N_EDGES  800000
#define IN_C     512
#define OUT_C    256
#define DEC_D    256
#define NUM_MASK 25000

__device__ __forceinline__ void atomAddF(float* p, float v) {
  __hip_atomic_fetch_add(p, v, __ATOMIC_RELAXED, __HIP_MEMORY_SCOPE_AGENT);
}

// ---------------- small utility kernels ----------------
__global__ void k_fill_f32(float* p, float v, int n) {
  int i = blockIdx.x * blockDim.x + threadIdx.x;
  if (i < n) p[i] = v;
}
__global__ void k_fill_i32(int* p, int v, int n) {
  int i = blockIdx.x * blockDim.x + threadIdx.x;
  if (i < n) p[i] = v;
}
__global__ void k_deg_edges(const int* __restrict__ dst, float* __restrict__ deg) {
  int e = blockIdx.x * blockDim.x + threadIdx.x;
  if (e < N_EDGES) atomAddF(&deg[dst[e]], 1.0f);
}
__global__ void k_rsqrt_inplace(float* p, int n) {
  int i = blockIdx.x * blockDim.x + threadIdx.x;
  if (i < n) p[i] = rsqrtf(p[i]);
}
__global__ void k_set_flags(const int* __restrict__ mi, int* __restrict__ flags) {
  int i = blockIdx.x * blockDim.x + threadIdx.x;
  if (i < NUM_MASK) flags[mi[i]] = 1;
}
// Bt[n*K + k] = (f16) W[k*N + n]   (W is K x N row-major)
__global__ void k_transpose_h(const float* __restrict__ W, _Float16* __restrict__ Bt,
                              int K, int N) {
  int i = blockIdx.x * blockDim.x + threadIdx.x;
  if (i < K * N) {
    int n = i / K, k = i % K;
    Bt[i] = (_Float16)W[k * N + n];
  }
}
// g[i,c] = t[i,c]*dinv[i]^2 + b[c]   (self-loop term + bias)
__global__ void k_agg_init(const float* __restrict__ t, const float* __restrict__ dinv,
                           const float* __restrict__ b, float* __restrict__ g, int C) {
  int idx = blockIdx.x * blockDim.x + threadIdx.x;
  if (idx < N_NODES * C) {
    int i = idx / C, c = idx % C;
    float di = dinv[i];
    g[idx] = t[idx] * di * di + b[c];
  }
}
// one wave32 per edge; edge data is wave-uniform -> scalar loads.
// lane handles 8 contiguous channels (C == 256).
__global__ void k_agg_edges_wave(const float* __restrict__ t, const int* __restrict__ src,
                                 const int* __restrict__ dst, const float* __restrict__ dinv,
                                 float* __restrict__ g) {
  const int wave = (blockIdx.x * blockDim.x + threadIdx.x) >> 5;
  const int lane = threadIdx.x & 31;
  if (wave >= N_EDGES) return;
  const int s = src[wave], d = dst[wave];
  const float coef = dinv[s] * dinv[d];
  const v4f* ts = (const v4f*)(t + (size_t)s * OUT_C) + lane * 2;
  float* gp = g + (size_t)d * OUT_C + lane * 8;
  v4f v0 = ts[0], v1 = ts[1];
#pragma unroll
  for (int j = 0; j < 4; ++j) atomAddF(gp + j, v0[j] * coef);
#pragma unroll
  for (int j = 0; j < 4; ++j) atomAddF(gp + 4 + j, v1[j] * coef);
}
__global__ void k_cvt_h(const float* __restrict__ g, _Float16* __restrict__ o,
                        int n, int relu) {
  int i = blockIdx.x * blockDim.x + threadIdx.x;
  if (i < n) {
    float v = g[i];
    if (relu) v = v > 0.0f ? v : 0.0f;
    o[i] = (_Float16)v;
  }
}
__global__ void k_copy_f32(const float* __restrict__ a, float* __restrict__ b, int n) {
  int i = blockIdx.x * blockDim.x + threadIdx.x;
  if (i < n) b[i] = a[i];
}
__global__ void k_idx_to_f32(const int* __restrict__ a, float* __restrict__ b, int n) {
  int i = blockIdx.x * blockDim.x + threadIdx.x;
  if (i < n) b[i] = (float)a[i];
}

// ---------------- strip-mined WMMA GEMM ----------------
// C(MxN) = A(MxK f16) * Bt(NxK f16)^T [+ bias].
// One wave32 owns a 16 x (16*NT) strip: 1 A fragment feeds NT back-to-back WMMAs
// per k-step (A traffic reduced NT x; B is small weight data, L0/L2-resident).
template <int K, int N, int NT, bool BIAS>
__global__ void k_wmma_gemm_h(const _Float16* __restrict__ A, const _Float16* __restrict__ Bt,
                              const float* __restrict__ bias, float* __restrict__ out) {
  const int lane = threadIdx.x & 31;
  const int wave = (blockIdx.x * blockDim.x + threadIdx.x) >> 5;
  constexpr int NG = N / (16 * NT);
  const int tiles = (N_NODES / 16) * NG;
  if (wave >= tiles) return;
  const int tm = wave / NG, tg = wave % NG;
  const int l16 = lane & 15, kh = lane >> 4;
  const _Float16* arow = A + (size_t)(tm * 16 + l16) * K + kh * 8;
  const _Float16* brow = Bt + (size_t)(tg * 16 * NT + l16) * K + kh * 8;
  v8f acc[NT];
#pragma unroll
  for (int j = 0; j < NT; ++j) acc[j] = (v8f){};
#pragma unroll 2
  for (int k0 = 0; k0 < K; k0 += 32) {
    v8h alo = *(const v8h*)(arow + k0);
    v8h ahi = *(const v8h*)(arow + k0 + 16);
    v16h a = __builtin_shufflevector(alo, ahi, 0,1,2,3,4,5,6,7,8,9,10,11,12,13,14,15);
#pragma unroll
    for (int j = 0; j < NT; ++j) {
      v8h blo = *(const v8h*)(brow + (size_t)j * 16 * K + k0);
      v8h bhi = *(const v8h*)(brow + (size_t)j * 16 * K + k0 + 16);
      v16h b = __builtin_shufflevector(blo, bhi, 0,1,2,3,4,5,6,7,8,9,10,11,12,13,14,15);
      acc[j] = __builtin_amdgcn_wmma_f32_16x16x32_f16(false, a, false, b, (short)0,
                                                      acc[j], false, false);
    }
  }
  const int orow = tm * 16 + 8 * kh;
#pragma unroll
  for (int j = 0; j < NT; ++j) {
    const int col = tg * 16 * NT + j * 16 + l16;
    const float bv = BIAS ? bias[col] : 0.0f;
#pragma unroll
    for (int v = 0; v < 8; ++v)
      out[(size_t)(orow + v) * N + col] = acc[j][v] + bv;
  }
}

// GEMM1: A is f32 x with masked rows replaced by token (via flags), cvt to f16 inline.
template <int NT>
__global__ void k_wmma_gemm_masked(const float* __restrict__ X, const int* __restrict__ flags,
                                   const float* __restrict__ token,
                                   const _Float16* __restrict__ Bt, float* __restrict__ out) {
  constexpr int K = IN_C, N = OUT_C;
  const int lane = threadIdx.x & 31;
  const int wave = (blockIdx.x * blockDim.x + threadIdx.x) >> 5;
  constexpr int NG = N / (16 * NT);
  const int tiles = (N_NODES / 16) * NG;
  if (wave >= tiles) return;
  const int tm = wave / NG, tg = wave % NG;
  const int l16 = lane & 15, kh = lane >> 4;
  const int row = tm * 16 + l16;
  const float* arow = (flags[row] ? token : X + (size_t)row * K) + kh * 8;
  const _Float16* brow = Bt + (size_t)(tg * 16 * NT + l16) * K + kh * 8;
  v8f acc[NT];
#pragma unroll
  for (int j = 0; j < NT; ++j) acc[j] = (v8f){};
#pragma unroll 2
  for (int k0 = 0; k0 < K; k0 += 32) {
    v4f a0 = *(const v4f*)(arow + k0);
    v4f a1 = *(const v4f*)(arow + k0 + 4);
    v4f a2 = *(const v4f*)(arow + k0 + 16);
    v4f a3 = *(const v4f*)(arow + k0 + 20);
    v16h a;
#pragma unroll
    for (int j = 0; j < 4; ++j) {
      a[j]      = (_Float16)a0[j];
      a[4 + j]  = (_Float16)a1[j];
      a[8 + j]  = (_Float16)a2[j];
      a[12 + j] = (_Float16)a3[j];
    }
#pragma unroll
    for (int j = 0; j < NT; ++j) {
      v8h blo = *(const v8h*)(brow + (size_t)j * 16 * K + k0);
      v8h bhi = *(const v8h*)(brow + (size_t)j * 16 * K + k0 + 16);
      v16h b = __builtin_shufflevector(blo, bhi, 0,1,2,3,4,5,6,7,8,9,10,11,12,13,14,15);
      acc[j] = __builtin_amdgcn_wmma_f32_16x16x32_f16(false, a, false, b, (short)0,
                                                      acc[j], false, false);
    }
  }
  const int orow = tm * 16 + 8 * kh;
#pragma unroll
  for (int j = 0; j < NT; ++j) {
    const int col = tg * 16 * NT + j * 16 + l16;
#pragma unroll
    for (int v = 0; v < 8; ++v)
      out[(size_t)(orow + v) * N + col] = acc[j][v];
  }
}

// ---------------- launcher ----------------
extern "C" void kernel_launch(void* const* d_in, const int* in_sizes, int n_in,
                              void* d_out, int out_size, void* d_ws, size_t ws_size,
                              hipStream_t stream) {
  (void)in_sizes; (void)n_in; (void)out_size; (void)ws_size;
  const float* x      = (const float*)d_in[0];
  const int*   ei     = (const int*)d_in[1];
  const int*   src    = ei;
  const int*   dst    = ei + N_EDGES;
  const int*   mi     = (const int*)d_in[2];
  const float* token  = (const float*)d_in[3];
  const float* enc_W  = (const float*)d_in[4];
  const float* enc_b  = (const float*)d_in[5];
  const float* dec_W  = (const float*)d_in[6];
  const float* dec_b  = (const float*)d_in[7];
  const float* mlp_W  = (const float*)d_in[8];
  const float* mlp_b  = (const float*)d_in[9];

  float* out       = (float*)d_out;
  float* out_recon = out;                                  // [50000,512]
  float* out_x     = out + (size_t)N_NODES * IN_C;         // [50000,512]
  float* out_mi    = out_x + (size_t)N_NODES * IN_C;       // [25000]

  // Stage big f32 temporaries inside d_out regions (finalized at the very end):
  float* t = out_recon;  // GEMM outputs [50000,256]
  float* g = out_x;      // aggregation outputs [50000,256]

  // workspace layout (~27 MB)
  char* ws = (char*)d_ws;
  size_t off = 0;
  auto alloc = [&](size_t bytes) -> void* {
    void* p = ws + off;
    off = (off + bytes + 255) & ~(size_t)255;
    return p;
  };
  float*     dinv  = (float*)alloc((size_t)N_NODES * 4);
  int*       flags = (int*)alloc((size_t)N_NODES * 4);
  _Float16*  enc_h = (_Float16*)alloc((size_t)IN_C * OUT_C * 2);
  _Float16*  dec_h = (_Float16*)alloc((size_t)OUT_C * DEC_D * 2);
  _Float16*  mlp_h = (_Float16*)alloc((size_t)DEC_D * IN_C * 2);
  _Float16*  ah    = (_Float16*)alloc((size_t)N_NODES * OUT_C * 2); // z_h then relu(h)_h

  const int T = 256;
  auto nb = [](long long n, int t) { return (int)((n + t - 1) / t); };

  // degrees + rsqrt, mask flags
  k_fill_f32<<<nb(N_NODES, T), T, 0, stream>>>(dinv, 1.0f, N_NODES);
  k_fill_i32<<<nb(N_NODES, T), T, 0, stream>>>(flags, 0, N_NODES);
  k_deg_edges<<<nb(N_EDGES, T), T, 0, stream>>>(dst, dinv);
  k_rsqrt_inplace<<<nb(N_NODES, T), T, 0, stream>>>(dinv, N_NODES);
  k_set_flags<<<nb(NUM_MASK, T), T, 0, stream>>>(mi, flags);

  // weight transposes to f16 (N x K)
  k_transpose_h<<<nb(IN_C * OUT_C, T), T, 0, stream>>>(enc_W, enc_h, IN_C, OUT_C);
  k_transpose_h<<<nb(OUT_C * DEC_D, T), T, 0, stream>>>(dec_W, dec_h, OUT_C, DEC_D);
  k_transpose_h<<<nb(DEC_D * IN_C, T), T, 0, stream>>>(mlp_W, mlp_h, DEC_D, IN_C);

  const int NC = N_NODES * OUT_C;                 // 12.8M
  const long long EWT = (long long)N_EDGES * 32;  // one wave per edge

  constexpr int NT = 8;  // 8 n-tiles per wave (64 accumulator VGPRs)

  // ---- encoder GCN ----
  {
    int waves = (N_NODES / 16) * (OUT_C / (16 * NT));  // 6250
    k_wmma_gemm_masked<NT><<<nb((long long)waves * 32, T), T, 0, stream>>>(x, flags, token,
                                                                           enc_h, t);
  }
  k_agg_init<<<nb(NC, T), T, 0, stream>>>(t, dinv, enc_b, g, OUT_C);
  k_agg_edges_wave<<<nb(EWT, T), T, 0, stream>>>(t, src, dst, dinv, g);
  k_cvt_h<<<nb(NC, T), T, 0, stream>>>(g, ah, NC, 0);          // z -> f16

  // ---- decoder GCN ----
  {
    int waves = (N_NODES / 16) * (DEC_D / (16 * NT));
    k_wmma_gemm_h<OUT_C, DEC_D, NT, false><<<nb((long long)waves * 32, T), T, 0, stream>>>(
        ah, dec_h, nullptr, t);
  }
  k_agg_init<<<nb(NC, T), T, 0, stream>>>(t, dinv, dec_b, g, DEC_D);
  k_agg_edges_wave<<<nb(EWT, T), T, 0, stream>>>(t, src, dst, dinv, g);
  k_cvt_h<<<nb(NC, T), T, 0, stream>>>(g, ah, NC, 1);          // relu(h) -> f16

  // ---- MLP head: x_recon = relu(h) @ mlp_W + mlp_b ----
  {
    int waves = (N_NODES / 16) * (IN_C / (16 * NT));   // 12500
    k_wmma_gemm_h<DEC_D, IN_C, NT, true><<<nb((long long)waves * 32, T), T, 0, stream>>>(
        ah, mlp_h, mlp_b, out_recon);
  }

  // ---- finalize remaining outputs (overwrites g region) ----
  k_copy_f32<<<nb(N_NODES * IN_C, T), T, 0, stream>>>(x, out_x, N_NODES * IN_C);
  k_idx_to_f32<<<nb(NUM_MASK, T), T, 0, stream>>>(mi, out_mi, NUM_MASK);
}
